// TokenSpikingTransformer_64407329571110
// MI455X (gfx1250) — matrix operand order repaired and checked by reading, not compiled, not run on hardware
//
#include <hip/hip_runtime.h>
#include <cstddef>

// ---------------- problem constants ----------------
#define TT    4
#define BB    32
#define CC    512
#define CHID  2048
#define NRE   196     // H*W
#define NPAD  208     // 13 tiles of 16
#define NHEAD 8
#define CHD   64
#define KC    128     // K chunk staged in LDS

typedef _Float16 vh16 __attribute__((ext_vector_type(16)));
typedef _Float16 vh8  __attribute__((ext_vector_type(8)));
typedef float    vf8  __attribute__((ext_vector_type(8)));

enum { MODE_QK = 0, MODE_PROJ = 1, MODE_MLP1 = 2, MODE_MLP2 = 3 };

// ---------------- weight f32 -> f16 ----------------
__global__ void cvt_f32_to_f16(const float* __restrict__ s, _Float16* __restrict__ d, int n) {
  for (int i = blockIdx.x * blockDim.x + threadIdx.x; i < n; i += gridDim.x * blockDim.x)
    d[i] = (_Float16)s[i];
}

// ------------- per-head q-spike sum + LIF(0.5) -------------
__global__ void attn_lif_kernel(const _Float16* __restrict__ qsp, float* __restrict__ attn) {
  int idx = blockIdx.x * blockDim.x + threadIdx.x;
  if (idx >= BB * NHEAD * NPAD) return;
  int n  = idx % NPAD;
  int hh = (idx / NPAD) % NHEAD;
  int b  = idx / (NPAD * NHEAD);
  float v = 0.f;
  for (int t = 0; t < TT; ++t) {
    int tb = t * BB + b;
    const _Float16* p = qsp + ((size_t)tb * CC + (size_t)hh * CHD) * NPAD + n;
    float s = 0.f;
    for (int c = 0; c < CHD; ++c) s += (float)p[(size_t)c * NPAD];
    float vv = 0.5f * (v + s);          // tau = 2
    float sp = (vv >= 0.5f) ? 1.f : 0.f;
    v = vv * (1.f - sp);
    attn[((size_t)tb * NHEAD + hh) * NPAD + n] = sp;
  }
}

// ------------- fused GEMM + BN + LIF over T (WMMA f16->f32) -------------
// D[o][n] = sum_c W[o][c] * X_t[c][n];  4 time-step accumulators -> LIF scan in regs.
template <int KDIM, int MODE>
__global__ __launch_bounds__(256) void gemm_bn_lif(
    const _Float16* __restrict__ W16,    // [M][KDIM] f16 row-major
    const float* __restrict__ bnS, const float* __restrict__ bnB,
    const float* __restrict__ bias,      // null for QK
    const float* __restrict__ xf32,      // QK/MLP1 input x [T,B,C,196]; MLP2: residual x
    const _Float16* __restrict__ spIn,   // PROJ: k_sp; MLP1: y_sp; MLP2: h1  [T,B,K,NPAD]
    const float* __restrict__ attn,      // PROJ only  [T,B,8,NPAD]
    const _Float16* __restrict__ ySp,    // MLP2 residual spikes [T,B,C,NPAD]
    _Float16* __restrict__ spOut,        // QK/PROJ/MLP1 spike out [T,B,M,NPAD]
    float* __restrict__ outF32)          // MLP2 final out [T,B,C,196]
{
  constexpr int MDIM = (MODE == MODE_MLP1) ? CHID : CC;
  constexpr int LROW = KC + 8;           // 272B row stride: 16B aligned, 4-bank stagger
  __shared__ _Float16 sB[TT * 16 * LROW];

  const int b    = blockIdx.z;
  const int n0   = blockIdx.y * 16;
  const int wave = threadIdx.x >> 5;
  const int lane = threadIdx.x & 31;
  const int l16  = lane & 15;
  const int hi   = lane >> 4;
  const int m0   = blockIdx.x * 128 + wave * 16;

  const int sn = threadIdx.x & 15;       // staging: column
  const int sk = threadIdx.x >> 4;       // staging: k row group

  vf8 acc[TT] = {};

  for (int k0 = 0; k0 < KDIM; k0 += KC) {
    __syncthreads();
    // ---- stage B tile [t][n][k] into LDS (with mode-specific fusion) ----
    #pragma unroll
    for (int t = 0; t < TT; ++t) {
      const int tb = t * BB + b;
      for (int kk = sk; kk < KC; kk += 16) {
        const int gk = k0 + kk;
        float v;
        if constexpr (MODE == MODE_QK || MODE == MODE_MLP1) {
          const int n = n0 + sn;
          float xv = (n < NRE) ? xf32[((size_t)tb * KDIM + gk) * NRE + n] : 0.f;
          if constexpr (MODE == MODE_MLP1)   // x_new = x + y_sp on the fly
            xv += (float)spIn[((size_t)tb * KDIM + gk) * NPAD + n0 + sn];
          v = xv;
        } else if constexpr (MODE == MODE_PROJ) {
          // x_one = attn_sp(head) * k_sp, fused into staging
          float kv = (float)spIn[((size_t)tb * KDIM + gk) * NPAD + n0 + sn];
          float av = attn[((size_t)tb * NHEAD + (gk >> 6)) * NPAD + n0 + sn];
          v = kv * av;
        } else { // MLP2: h1 spikes
          v = (float)spIn[((size_t)tb * KDIM + gk) * NPAD + n0 + sn];
        }
        sB[(t * 16 + sn) * LROW + kk] = (_Float16)v;
      }
    }
    __syncthreads();

    // ---- WMMA over the chunk; A fragment reused across 4 time steps ----
    #pragma unroll
    for (int kk = 0; kk < KC; kk += 32) {
      // A: lane (m=l16, hi): K = base + hi*16 + 0..15 contiguous, 2x16B from global
      const _Float16* wp = W16 + (size_t)(m0 + l16) * KDIM + (k0 + kk) + hi * 16;
      vh8 a0 = *(const vh8*)wp;
      vh8 a1 = *(const vh8*)(wp + 8);
      vh16 A = __builtin_shufflevector(a0, a1, 0,1,2,3,4,5,6,7,8,9,10,11,12,13,14,15);
      #pragma unroll
      for (int t = 0; t < TT; ++t) {
        // B: lane (n=l16, hi): K = kk + hi*16 + 0..15 contiguous in LDS
        const _Float16* bp = &sB[(t * 16 + l16) * LROW + kk + hi * 16];
        vh8 b0 = *(const vh8*)bp;
        vh8 b1 = *(const vh8*)(bp + 8);
        vh16 Bf = __builtin_shufflevector(b0, b1, 0,1,2,3,4,5,6,7,8,9,10,11,12,13,14,15);
        acc[t] = __builtin_amdgcn_wmma_f32_16x16x32_f16(
            false, A, false, Bf, (short)0, acc[t], false, false);
      }
    }
  }

  // ---- fold bias into BN affine; LIF scan over T in registers ----
  float es[8], eb[8];
  #pragma unroll
  for (int r = 0; r < 8; ++r) {
    const int o = m0 + r + hi * 8;       // D row = r + 8*hi
    float s = bnS[o];
    es[r] = s;
    float e = bnB[o];
    if constexpr (MODE != MODE_QK) e += s * bias[o];
    eb[r] = e;
  }
  float v[8];
  #pragma unroll
  for (int r = 0; r < 8; ++r) v[r] = 0.f;

  #pragma unroll
  for (int t = 0; t < TT; ++t) {
    const int tb = t * BB + b;
    #pragma unroll
    for (int r = 0; r < 8; ++r) {
      const int o = m0 + r + hi * 8;
      float pre = acc[t][r] * es[r] + eb[r];
      float vv  = 0.5f * (v[r] + pre);   // v += (x - v)/tau, tau=2
      float sp  = (vv >= 1.0f) ? 1.f : 0.f;
      v[r] = vv * (1.f - sp);            // hard reset
      if constexpr (MODE == MODE_MLP2) {
        const int n = n0 + l16;
        if (n < NRE) {
          size_t oi = ((size_t)tb * CC + o) * NRE + n;
          outF32[oi] = xf32[oi]
                     + (float)ySp[((size_t)tb * CC + o) * NPAD + n0 + l16]
                     + sp;               // out = x + y_sp + h2_sp
        }
      } else {
        spOut[((size_t)tb * MDIM + o) * NPAD + n0 + l16] = (_Float16)sp;
      }
    }
  }
}

// ---------------- host orchestration ----------------
extern "C" void kernel_launch(void* const* d_in, const int* in_sizes, int n_in,
                              void* d_out, int out_size, void* d_ws, size_t ws_size,
                              hipStream_t stream) {
  (void)in_sizes; (void)n_in; (void)out_size; (void)ws_size;
  const float* x     = (const float*)d_in[0];
  const float* q_w   = (const float*)d_in[1];
  const float* q_s   = (const float*)d_in[2];
  const float* q_b   = (const float*)d_in[3];
  const float* k_w   = (const float*)d_in[4];
  const float* k_s   = (const float*)d_in[5];
  const float* k_b   = (const float*)d_in[6];
  const float* p_w   = (const float*)d_in[7];
  const float* p_bi  = (const float*)d_in[8];
  const float* p_s   = (const float*)d_in[9];
  const float* p_b   = (const float*)d_in[10];
  const float* m1_w  = (const float*)d_in[11];
  const float* m1_bi = (const float*)d_in[12];
  const float* m1_s  = (const float*)d_in[13];
  const float* m1_b  = (const float*)d_in[14];
  const float* m2_w  = (const float*)d_in[15];
  const float* m2_bi = (const float*)d_in[16];
  const float* m2_s  = (const float*)d_in[17];
  const float* m2_b  = (const float*)d_in[18];

  char* ws = (char*)d_ws;
  const size_t SZ_W_CC = (size_t)CC * CC * sizeof(_Float16);                 // 512 KB
  const size_t SZ_W_CH = (size_t)CHID * CC * sizeof(_Float16);               // 2 MB
  const size_t SZ_SP_C = (size_t)TT * BB * CC * NPAD * sizeof(_Float16);     // 26 MB
  const size_t SZ_H1   = (size_t)TT * BB * CHID * NPAD * sizeof(_Float16);   // 104 MB

  _Float16* wq  = (_Float16*)(ws);
  _Float16* wk  = (_Float16*)(ws + SZ_W_CC);
  _Float16* wpj = (_Float16*)(ws + 2 * SZ_W_CC);
  _Float16* wm1 = (_Float16*)(ws + 3 * SZ_W_CC);
  _Float16* wm2 = (_Float16*)(ws + 3 * SZ_W_CC + SZ_W_CH);
  char* dyn = ws + 3 * SZ_W_CC + 2 * SZ_W_CH;
  _Float16* h1  = (_Float16*)dyn;                 // [T,B,Ch,NPAD], aliases qsp+ksp (dead by then)
  _Float16* qsp = (_Float16*)dyn;                 // [T,B,C,NPAD]
  _Float16* ksp = (_Float16*)(dyn + SZ_SP_C);     // [T,B,C,NPAD]
  _Float16* ysp = (_Float16*)(dyn + SZ_H1);       // [T,B,C,NPAD]
  float*    att = (float*)(dyn + SZ_H1 + SZ_SP_C);// [T,B,8,NPAD]
  float*    out = (float*)d_out;

  dim3 blk(256);
  cvt_f32_to_f16<<<dim3(512),  blk, 0, stream>>>(q_w,  wq,  CC * CC);
  cvt_f32_to_f16<<<dim3(512),  blk, 0, stream>>>(k_w,  wk,  CC * CC);
  cvt_f32_to_f16<<<dim3(512),  blk, 0, stream>>>(p_w,  wpj, CC * CC);
  cvt_f32_to_f16<<<dim3(2048), blk, 0, stream>>>(m1_w, wm1, CHID * CC);
  cvt_f32_to_f16<<<dim3(2048), blk, 0, stream>>>(m2_w, wm2, CC * CHID);

  dim3 gC(CC / 128, NPAD / 16, BB);      // 4 x 13 x 32
  dim3 gH(CHID / 128, NPAD / 16, BB);    // 16 x 13 x 32

  gemm_bn_lif<CC,   MODE_QK  ><<<gC, blk, 0, stream>>>(wq,  q_s,  q_b,  nullptr, x,       nullptr, nullptr, nullptr, qsp, nullptr);
  gemm_bn_lif<CC,   MODE_QK  ><<<gC, blk, 0, stream>>>(wk,  k_s,  k_b,  nullptr, x,       nullptr, nullptr, nullptr, ksp, nullptr);
  attn_lif_kernel<<<dim3((BB * NHEAD * NPAD + 255) / 256), blk, 0, stream>>>(qsp, att);
  gemm_bn_lif<CC,   MODE_PROJ><<<gC, blk, 0, stream>>>(wpj, p_s,  p_b,  p_bi,   nullptr, ksp,     att,     nullptr, ysp, nullptr);
  gemm_bn_lif<CC,   MODE_MLP1><<<gH, blk, 0, stream>>>(wm1, m1_s, m1_b, m1_bi,  x,       ysp,     nullptr, nullptr, h1,  nullptr);
  gemm_bn_lif<CHID, MODE_MLP2><<<gC, blk, 0, stream>>>(wm2, m2_s, m2_b, m2_bi,  x,       h1,      nullptr, ysp,     nullptr, out);
}
